// Model_76501957476616
// MI455X (gfx1250) — matrix operand order repaired and checked by reading, not compiled, not run on hardware
//
#include <hip/hip_runtime.h>
#include <hip/hip_bf16.h>
#include <math.h>

// ---------------------------------------------------------------------------
// MI455X (gfx1250) implementation of the GIN contrastive model.
// GEMMs use v_wmma_f32_16x16x32_bf16 with operands pre-packed into CDNA5
// wave32 fragment layouts; 16-row x 64-col tile per wave (spill-free),
// register double-buffered K loop (4 WMMAs per K-step).
// ---------------------------------------------------------------------------

#define EMB       300
#define EMBP      320     // padded feature stride
#define HID       600
#define HIDP      640     // padded hidden stride
#define N_NODES   100000
#define N_EDGES   200000
#define N_FRAGS   10000
#define N_SEDGES  20000
#define N_SG      2048
#define HALF_SG   1024
#define BN_EPS    1e-5f
#define INV_TEMP  25.0f   // 1/0.04

typedef __attribute__((ext_vector_type(16))) __bf16 v16bf;
typedef __attribute__((ext_vector_type(8)))  float  v8f;

__device__ __forceinline__ unsigned short f32_to_bf16(float f) {
    unsigned int u = __float_as_uint(f);
    u += 0x7FFFu + ((u >> 16) & 1u);          // round-to-nearest-even
    return (unsigned short)(u >> 16);
}

// ---------------------------------------------------------------------------
// Elementwise / gather / scatter kernels
// ---------------------------------------------------------------------------

__global__ void k_zero(float* p, int n) {
    int i = blockIdx.x * 256 + threadIdx.x;
    if (i < n) p[i] = 0.0f;
}

// h[i,f] = atom_e1[x0,f] + atom_e2[x1,f]   (pad cols -> 0)
__global__ void k_embed(const int* __restrict__ x,
                        const float* __restrict__ e1,
                        const float* __restrict__ e2,
                        float* __restrict__ h) {
    int i = blockIdx.x, f = threadIdx.x;
    float v = 0.0f;
    if (f < EMB) {
        int a0 = x[2 * i], a1 = x[2 * i + 1];
        v = e1[(size_t)a0 * EMB + f] + e2[(size_t)a1 * EMB + f];
    }
    h[(size_t)i * EMBP + f] = v;
}

// agg[i,f] = h[i,f] + self_e[f]  (self-loop term; pad cols -> 0)
__global__ void k_self(const float* __restrict__ h,
                       const float* __restrict__ se1,
                       const float* __restrict__ se2,
                       float* __restrict__ agg) {
    int i = blockIdx.x, f = threadIdx.x;
    float v = 0.0f;
    if (f < EMB) v = h[(size_t)i * EMBP + f] + se1[f] + se2[f];
    agg[(size_t)i * EMBP + f] = v;
}

// agg[dst,f] += h[src,f] + ee1[a0,f] + ee2[a1,f]
__global__ void k_scatter(const float* __restrict__ h,
                          const int* __restrict__ ei,      // (2, E)
                          const int* __restrict__ ea,      // (E, 2)
                          const float* __restrict__ ee1,
                          const float* __restrict__ ee2,
                          float* __restrict__ agg, int E) {
    int e = blockIdx.x, f = threadIdx.x;
    if (f >= EMB) return;
    int src = ei[e], dst = ei[E + e];
    int a0 = ea[2 * e], a1 = ea[2 * e + 1];
    float msg = h[(size_t)src * EMBP + f] +
                ee1[(size_t)a0 * EMB + f] + ee2[(size_t)a1 * EMB + f];
    atomicAdd(&agg[(size_t)dst * EMBP + f], msg);
}

// ---------------------------------------------------------------------------
// Fragment packing (CDNA5 wave32 WMMA layouts, bf16)
//
// B-fragment layout: frag[((kt*NT + nt)*32 + lane)*16 + j]
//   lane 0-15 : N = nt*16 + lane,      K = kt*32 + (j&7) + 16*(j>>3)
//   lane 16-31: N = nt*16 + lane - 16, K = kt*32 + 8 + (j&7) + 16*(j>>3)
//
// A-fragment layout: frag[((mt*KT + kt)*32 + lane)*16 + j]
//   row = mt*16 + (lane&15), hi = lane>>4
//   j<8 : K = kt*32 + hi*8 + j
//   j>=8: K = kt*32 + 16 + hi*8 + (j-8)
// ---------------------------------------------------------------------------

__global__ void k_pack_w(const float* __restrict__ W, unsigned short* __restrict__ out,
                         int Kreal, int Nreal, int NT, int total) {
    int idx = blockIdx.x * 256 + threadIdx.x;
    if (idx >= total) return;
    int j    = idx & 15;
    int lane = (idx >> 4) & 31;
    int nt   = (idx >> 9) % NT;
    int kt   = (idx >> 9) / NT;
    int k = kt * 32 + ((lane >> 4) << 3) + (j & 7) + ((j >> 3) << 4);
    int n = nt * 16 + (lane & 15);
    float v = (k < Kreal && n < Nreal) ? W[(size_t)k * Nreal + n] : 0.0f;
    out[idx] = f32_to_bf16(v);
}

// pack B = f1^T where f1 = feats[HALF_SG ..): B[k][n] = feats[HALF_SG+n][k]
__global__ void k_pack_feats(const float* __restrict__ feats,
                             unsigned short* __restrict__ out, int NT, int total) {
    int idx = blockIdx.x * 256 + threadIdx.x;
    if (idx >= total) return;
    int j    = idx & 15;
    int lane = (idx >> 4) & 31;
    int nt   = (idx >> 9) % NT;
    int kt   = (idx >> 9) / NT;
    int k = kt * 32 + ((lane >> 4) << 3) + (j & 7) + ((j >> 3) << 4); // < EMBP (pad is 0)
    int n = nt * 16 + (lane & 15);
    out[idx] = f32_to_bf16(feats[(size_t)(HALF_SG + n) * EMBP + k]);
}

// pack f32 row-major [Mreal x lda] into A-fragment layout
__global__ void k_pack_a(const float* __restrict__ A, unsigned short* __restrict__ out,
                         int KT, int lda, int Mreal, int total) {
    int idx = blockIdx.x * 256 + threadIdx.x;
    if (idx >= total) return;
    int j    = idx & 15;
    int lane = (idx >> 4) & 31;
    int kt   = (idx >> 9) % KT;
    int mt   = (idx >> 9) / KT;
    int hi   = lane >> 4;
    int k    = kt * 32 + (j < 8 ? hi * 8 + j : 16 + hi * 8 + (j - 8));
    int row  = mt * 16 + (lane & 15);
    float v  = (row < Mreal) ? A[(size_t)row * lda + k] : 0.0f;
    out[idx] = f32_to_bf16(v);
}

// ---------------------------------------------------------------------------
// WMMA GEMM:  Out[M x Npad] = act( Apack @ Bpack * scale + bias )
// One wave per block computes a 16x64 C tile. K loop double-buffered:
// per kt: 1 A-fragment load + 4 B-fragment loads (32B each) + 4 WMMAs.
// OUT_MODE 0: f32 row-major (stride p_ldo)
// OUT_MODE 1: bf16 A-fragment layout for the next GEMM (K-tiles = p_kt2)
// grid = (M/16, Npad/64), block = 32.
// ---------------------------------------------------------------------------

template <bool RELU, int OUT_MODE>
__global__ void k_gemm(const unsigned short* __restrict__ Ap,
                       const unsigned short* __restrict__ Bp,
                       const float* __restrict__ bias, float scale,
                       void* __restrict__ Outv,
                       int Nreal, int Npad, int KT, int p_ldo, int p_kt2) {
    int lane = threadIdx.x;
    int mt   = blockIdx.x;
    int nb   = blockIdx.y;
    int NT   = Npad >> 4;
    int hi   = lane >> 4;

    const unsigned short* aP = Ap + ((size_t)mt * KT) * 512 + lane * 16;
    const unsigned short* bP = Bp + ((size_t)nb * 4) * 512 + lane * 16;
    const size_t bStep = (size_t)NT * 512;

    v8f acc0 = {}, acc1 = {}, acc2 = {}, acc3 = {};

    v16bf a  = *(const v16bf*)aP;
    v16bf b0 = *(const v16bf*)(bP);
    v16bf b1 = *(const v16bf*)(bP + 512);
    v16bf b2 = *(const v16bf*)(bP + 1024);
    v16bf b3 = *(const v16bf*)(bP + 1536);

    for (int kt = 0; kt < KT; ++kt) {
        v16bf an = a, bn0 = b0, bn1 = b1, bn2 = b2, bn3 = b3;
        if (kt + 1 < KT) {
            aP += 512;
            bP += bStep;
            __builtin_prefetch(aP + 1024, 0, 3);
            an  = *(const v16bf*)aP;
            bn0 = *(const v16bf*)(bP);
            bn1 = *(const v16bf*)(bP + 512);
            bn2 = *(const v16bf*)(bP + 1024);
            bn3 = *(const v16bf*)(bP + 1536);
        }
        acc0 = __builtin_amdgcn_wmma_f32_16x16x32_bf16(false, a, false, b0,
                                                       (short)0, acc0, false, false);
        acc1 = __builtin_amdgcn_wmma_f32_16x16x32_bf16(false, a, false, b1,
                                                       (short)0, acc1, false, false);
        acc2 = __builtin_amdgcn_wmma_f32_16x16x32_bf16(false, a, false, b2,
                                                       (short)0, acc2, false, false);
        acc3 = __builtin_amdgcn_wmma_f32_16x16x32_bf16(false, a, false, b3,
                                                       (short)0, acc3, false, false);
        a = an; b0 = bn0; b1 = bn1; b2 = bn2; b3 = bn3;
    }

    // C layout: VGPR g -> row = mt*16 + hi*8 + g, col = nb*64 + t*16 + (lane&15)
    int r0 = mt * 16 + hi * 8;
    int c0 = nb * 64 + (lane & 15);
    v8f accs[4] = {acc0, acc1, acc2, acc3};
#pragma unroll
    for (int t = 0; t < 4; ++t) {
        int col = c0 + t * 16;
        float bv = (bias != nullptr && col < Nreal) ? bias[col] : 0.0f;
#pragma unroll
        for (int g = 0; g < 8; ++g) {
            float v = accs[t][g] * scale + bv;
            if (RELU) v = fmaxf(v, 0.0f);
            int row = r0 + g;
            if (OUT_MODE == 0) {
                ((float*)Outv)[(size_t)row * p_ldo + col] = v;
            } else {
                // scatter into A-fragment layout of the next GEMM
                int kt2 = col >> 5;
                int kk  = col & 31;
                int hi2 = (kk >> 3) & 1;                    // K-half select
                int j2  = (kk & 7) + ((kk >> 4) << 3);      // fragment element
                int ln2 = hi2 * 16 + (row & 15);
                size_t o = ((size_t)((row >> 4) * p_kt2 + kt2) * 32 + ln2) * 16 + j2;
                ((unsigned short*)Outv)[o] = f32_to_bf16(v);
            }
        }
    }
}

// ---------------------------------------------------------------------------
// BatchNorm (population stats over M rows) — coalesced partial sums
// ---------------------------------------------------------------------------

// each block accumulates a 128-row chunk; thread = feature (contiguous loads)
__global__ void k_bn_partial(const float* __restrict__ h, float* __restrict__ sum,
                             float* __restrict__ ssum, int M) {
    int f  = threadIdx.x;           // 0..EMBP-1
    int r0 = blockIdx.x * 128;
    int r1 = r0 + 128 < M ? r0 + 128 : M;
    if (f >= EMB) return;
    float s = 0.0f, ss = 0.0f;
    for (int i = r0; i < r1; ++i) {
        float v = h[(size_t)i * EMBP + f];
        s += v; ss += v * v;
    }
    atomicAdd(&sum[f], s);
    atomicAdd(&ssum[f], ss);
}

__global__ void k_bn_finalize(const float* __restrict__ sum, const float* __restrict__ ssum,
                              const float* __restrict__ g, const float* __restrict__ b,
                              float* __restrict__ sc, float* __restrict__ sh, int M) {
    int f = blockIdx.x * 256 + threadIdx.x;
    if (f >= EMB) return;
    float mu  = sum[f] / (float)M;
    float var = ssum[f] / (float)M - mu * mu;
    float inv = rsqrtf(var + BN_EPS);
    float scale = g[f] * inv;
    sc[f] = scale;
    sh[f] = b[f] - mu * scale;
}

__global__ void k_bn_apply(float* __restrict__ h, const float* __restrict__ sc,
                           const float* __restrict__ sh, int relu) {
    int i = blockIdx.x, f = threadIdx.x;
    if (f >= EMB) return;
    size_t o = (size_t)i * EMBP + f;
    float v = h[o] * sc[f] + sh[f];
    if (relu) v = fmaxf(v, 0.0f);
    h[o] = v;
}

// ---------------------------------------------------------------------------
// Segment mean pooling + mask / normalize
// ---------------------------------------------------------------------------

__global__ void k_count(const int* __restrict__ seg, float* __restrict__ cnt, int n) {
    int i = blockIdx.x * 256 + threadIdx.x;
    if (i < n) atomicAdd(&cnt[seg[i]], 1.0f);
}

__global__ void k_pool_sum(const float* __restrict__ h, const int* __restrict__ seg,
                           float* __restrict__ sum, int n) {
    int i = blockIdx.x, f = threadIdx.x;
    if (i >= n || f >= EMB) return;
    atomicAdd(&sum[(size_t)seg[i] * EMBP + f], h[(size_t)i * EMBP + f]);
}

__global__ void k_frag_final(float* __restrict__ hs, const float* __restrict__ sum,
                             const float* __restrict__ cnt,
                             const unsigned char* __restrict__ mask,
                             const float* __restrict__ memb) {
    int g = blockIdx.x, f = threadIdx.x;
    float v = 0.0f;
    if (f < EMB)
        v = mask[g] ? memb[f]
                    : sum[(size_t)g * EMBP + f] / fmaxf(cnt[g], 1.0f);
    hs[(size_t)g * EMBP + f] = v;
}

__global__ void k_normalize(float* __restrict__ feats, const float* __restrict__ sum,
                            const float* __restrict__ cnt) {
    int g = blockIdx.x, f = threadIdx.x;   // block = EMBP
    float v = 0.0f;
    if (f < EMB) v = sum[(size_t)g * EMBP + f] / fmaxf(cnt[g], 1.0f);
    __shared__ float red[EMBP];
    __shared__ float nrm;
    red[f] = v * v;
    __syncthreads();
    if (f == 0) {
        float s = 0.0f;
        for (int k = 0; k < EMB; ++k) s += red[k];
        nrm = fmaxf(sqrtf(s), 1e-12f);
    }
    __syncthreads();
    feats[(size_t)g * EMBP + f] = (f < EMB) ? v / nrm : 0.0f;
}

__global__ void k_labels(float* __restrict__ out) {
    int i = blockIdx.x * 256 + threadIdx.x;
    if (i < HALF_SG) out[(size_t)HALF_SG * HALF_SG + i] = (float)i;
}

// ---------------------------------------------------------------------------
// Host launcher
// ---------------------------------------------------------------------------

extern "C" void kernel_launch(void* const* d_in, const int* in_sizes, int n_in,
                              void* d_out, int out_size, void* d_ws, size_t ws_size,
                              hipStream_t stream) {
    (void)in_sizes; (void)n_in; (void)out_size; (void)ws_size;

    const int*  x        = (const int*)d_in[0];
    const int*  ei       = (const int*)d_in[1];
    const int*  ea       = (const int*)d_in[2];
    const int*  batch    = (const int*)d_in[3];
    const unsigned char* smask = (const unsigned char*)d_in[4];
    const int*  sei      = (const int*)d_in[5];
    const int*  sea      = (const int*)d_in[6];
    const int*  sbatch   = (const int*)d_in[7];
    const float* atom_e1 = (const float*)d_in[9];
    const float* atom_e2 = (const float*)d_in[10];
    const float* edge_e1 = (const float*)d_in[11];  // (5,6,300)
    const float* edge_e2 = (const float*)d_in[12];  // (5,3,300)
    const float* W1      = (const float*)d_in[13];  // (5,300,600)
    const float* b1      = (const float*)d_in[14];
    const float* W2      = (const float*)d_in[15];  // (5,600,300)
    const float* b2      = (const float*)d_in[16];
    const float* bn_g    = (const float*)d_in[17];
    const float* bn_b    = (const float*)d_in[18];
    const float* s_e1    = (const float*)d_in[19];  // (6,300)
    const float* s_e2    = (const float*)d_in[20];  // (3,300)
    const float* sW1     = (const float*)d_in[21];
    const float* sb1     = (const float*)d_in[22];
    const float* sW2     = (const float*)d_in[23];
    const float* sb2     = (const float*)d_in[24];
    const float* memb    = (const float*)d_in[25];

    char* ws = (char*)d_ws;
    size_t off = 0;
    auto alloc = [&](size_t bytes) -> char* {
        char* p = ws + off;
        off += (bytes + 255) & ~(size_t)255;
        return p;
    };

    float*          h     = (float*)alloc((size_t)N_NODES * EMBP * 4);
    float*          agg   = (float*)alloc((size_t)N_NODES * EMBP * 4);
    unsigned short* apack = (unsigned short*)alloc((size_t)N_NODES * EMBP * 2);
    unsigned short* t1f   = (unsigned short*)alloc((size_t)N_NODES * HIDP * 2);
    unsigned short* bp1   = (unsigned short*)alloc(204800 * 2);
    unsigned short* bp2   = (unsigned short*)alloc(204800 * 2);
    float*          bnred = (float*)alloc(2 * EMBP * 4);     // sum | sumsq
    float*          bnsc  = (float*)alloc(EMBP * 4);
    float*          bnsh  = (float*)alloc(EMBP * 4);
    float*          fsum  = (float*)alloc((size_t)N_FRAGS * EMBP * 4);
    float*          fcnt  = (float*)alloc((size_t)N_FRAGS * 4);
    float*          hs    = (float*)alloc((size_t)N_FRAGS * EMBP * 4);
    float*          aggs  = (float*)alloc((size_t)N_FRAGS * EMBP * 4);
    float*          hs2   = (float*)alloc((size_t)N_FRAGS * EMBP * 4);
    float*          gsum  = (float*)alloc((size_t)N_SG * EMBP * 4);
    float*          gcnt  = (float*)alloc((size_t)N_SG * 4);
    float*          feats = (float*)alloc((size_t)N_SG * EMBP * 4);
    unsigned short* bpF   = (unsigned short*)alloc(327680 * 2);
    unsigned short* apF   = (unsigned short*)alloc((size_t)HALF_SG * EMBP * 2);

    const int KT_E = EMBP / 32;   // 10
    const int KT_H = HIDP / 32;   // 20

    // 1) node embedding
    k_embed<<<N_NODES, EMBP, 0, stream>>>(x, atom_e1, atom_e2, h);

    // 2) five GIN layers
    for (int l = 0; l < 5; ++l) {
        const float* ee1 = edge_e1 + (size_t)l * 6 * EMB;
        const float* ee2 = edge_e2 + (size_t)l * 3 * EMB;
        const float* se1 = ee1 + (size_t)4 * EMB;   // SELF_LOOP_BOND = 4
        const float* se2 = ee2;                     // dir 0
        const float* W1l = W1 + (size_t)l * EMB * HID;
        const float* b1l = b1 + (size_t)l * HID;
        const float* W2l = W2 + (size_t)l * HID * EMB;
        const float* b2l = b2 + (size_t)l * EMB;

        k_pack_w<<<(204800 + 255) / 256, 256, 0, stream>>>(W1l, bp1, EMB, HID, HIDP / 16, 204800);
        k_pack_w<<<(204800 + 255) / 256, 256, 0, stream>>>(W2l, bp2, HID, EMB, EMBP / 16, 204800);

        k_self<<<N_NODES, EMBP, 0, stream>>>(h, se1, se2, agg);
        k_scatter<<<N_EDGES, EMBP, 0, stream>>>(h, ei, ea, ee1, ee2, agg, N_EDGES);

        // pack agg -> A fragments (bf16)
        {
            int total = (N_NODES / 16) * KT_E * 512;
            k_pack_a<<<(total + 255) / 256, 256, 0, stream>>>(agg, apack, KT_E, EMBP,
                                                              N_NODES, total);
        }

        // t1f = relu(agg @ W1 + b1), stored directly as A-fragments for GEMM2
        k_gemm<true, 1><<<dim3(N_NODES / 16, HIDP / 64), 32, 0, stream>>>(
            apack, bp1, b1l, 1.0f, t1f, HID, HIDP, KT_E, 0, KT_H);
        // h = t1f @ W2 + b2  (f32 row-major)
        k_gemm<false, 0><<<dim3(N_NODES / 16, EMBP / 64), 32, 0, stream>>>(
            t1f, bp2, b2l, 1.0f, h, EMB, EMBP, KT_H, EMBP, 0);

        // batchnorm (+ relu except last layer)
        k_zero<<<(2 * EMBP + 255) / 256, 256, 0, stream>>>(bnred, 2 * EMBP);
        k_bn_partial<<<(N_NODES + 127) / 128, EMBP, 0, stream>>>(h, bnred, bnred + EMBP,
                                                                 N_NODES);
        k_bn_finalize<<<(EMB + 255) / 256, 256, 0, stream>>>(
            bnred, bnred + EMBP, bn_g + (size_t)l * EMB, bn_b + (size_t)l * EMB,
            bnsc, bnsh, N_NODES);
        k_bn_apply<<<N_NODES, EMBP, 0, stream>>>(h, bnsc, bnsh, (l < 4) ? 1 : 0);
    }

    // 3) fragment mean-pool + mask
    k_zero<<<((N_FRAGS * EMBP) + 255) / 256, 256, 0, stream>>>(fsum, N_FRAGS * EMBP);
    k_zero<<<(N_FRAGS + 255) / 256, 256, 0, stream>>>(fcnt, N_FRAGS);
    k_count<<<(N_NODES + 255) / 256, 256, 0, stream>>>(batch, fcnt, N_NODES);
    k_pool_sum<<<N_NODES, EMBP, 0, stream>>>(h, batch, fsum, N_NODES);
    k_frag_final<<<N_FRAGS, EMBP, 0, stream>>>(hs, fsum, fcnt, smask, memb);

    // 4) super GIN conv
    k_pack_w<<<(204800 + 255) / 256, 256, 0, stream>>>(sW1, bp1, EMB, HID, HIDP / 16, 204800);
    k_pack_w<<<(204800 + 255) / 256, 256, 0, stream>>>(sW2, bp2, HID, EMB, EMBP / 16, 204800);
    k_self<<<N_FRAGS, EMBP, 0, stream>>>(hs, s_e1 + (size_t)4 * EMB, s_e2, aggs);
    k_scatter<<<N_SEDGES, EMBP, 0, stream>>>(hs, sei, sea, s_e1, s_e2, aggs, N_SEDGES);
    {
        int total = (N_FRAGS / 16) * KT_E * 512;
        k_pack_a<<<(total + 255) / 256, 256, 0, stream>>>(aggs, apack, KT_E, EMBP,
                                                          N_FRAGS, total);
    }
    k_gemm<true, 1><<<dim3(N_FRAGS / 16, HIDP / 64), 32, 0, stream>>>(
        apack, bp1, sb1, 1.0f, t1f, HID, HIDP, KT_E, 0, KT_H);
    k_gemm<false, 0><<<dim3(N_FRAGS / 16, EMBP / 64), 32, 0, stream>>>(
        t1f, bp2, sb2, 1.0f, hs2, EMB, EMBP, KT_H, EMBP, 0);

    // 5) super-graph mean-pool + L2 normalize
    k_zero<<<((N_SG * EMBP) + 255) / 256, 256, 0, stream>>>(gsum, N_SG * EMBP);
    k_zero<<<(N_SG + 255) / 256, 256, 0, stream>>>(gcnt, N_SG);
    k_count<<<(N_FRAGS + 255) / 256, 256, 0, stream>>>(sbatch, gcnt, N_FRAGS);
    k_pool_sum<<<N_FRAGS, EMBP, 0, stream>>>(hs2, sbatch, gsum, N_FRAGS);
    k_normalize<<<N_SG, EMBP, 0, stream>>>(feats, gsum, gcnt);

    // 6) logits = f0 @ f1^T / TEMP -> d_out ; labels appended
    k_pack_feats<<<(327680 + 255) / 256, 256, 0, stream>>>(feats, bpF, HALF_SG / 16, 327680);
    {
        int total = (HALF_SG / 16) * KT_E * 512;
        k_pack_a<<<(total + 255) / 256, 256, 0, stream>>>(feats, apF, KT_E, EMBP,
                                                          HALF_SG, total);
    }
    k_gemm<false, 0><<<dim3(HALF_SG / 16, HALF_SG / 64), 32, 0, stream>>>(
        apF, bpF, nullptr, INV_TEMP, (float*)d_out, HALF_SG, HALF_SG, KT_E, HALF_SG, 0);
    k_labels<<<(HALF_SG + 255) / 256, 256, 0, stream>>>((float*)d_out);
}
